// TransformerBlock_31653908971698
// MI455X (gfx1250) — compile-verified, matrix operand order checked by reading, not run on hardware
//
#include <hip/hip_runtime.h>

// ---------------- problem constants ----------------
#define BATCH   2
#define SEQ     2048
#define DM      1024
#define HEADS   16
#define DK      64
#define DFF     4096
#define MROWS   (BATCH * SEQ)       // 4096
#define EPSV    1e-5f
#define THETA_F 10000.0f

// ---------------- types ----------------
typedef __bf16 bf16;
typedef __attribute__((ext_vector_type(16))) __bf16 bf16x16;
typedef __attribute__((ext_vector_type(8)))  __bf16 bf16x8;
typedef __attribute__((ext_vector_type(8)))  float  f32x8;
typedef __attribute__((ext_vector_type(4)))  unsigned int u32x4;

// Load one 16x32 bf16 WMMA fragment (A or B operand) from LDS.
// rowPtr points at the start of this lane's row (16-byte aligned, row stride
// is a multiple of 8 elements). kh = lane>>4 selects the K-half per the ISA
// 16-bit A-matrix layout: half 0 -> K {0..7, 16..23}, half 1 -> K {8..15, 24..31}.
__device__ __forceinline__ bf16x16 frag_ld(const bf16* rowPtr, int kh) {
    bf16x8 lo = *(const bf16x8*)(rowPtr + kh * 8);
    bf16x8 hi = *(const bf16x8*)(rowPtr + 16 + kh * 8);
    return __builtin_shufflevector(lo, hi, 0,1,2,3,4,5,6,7,8,9,10,11,12,13,14,15);
}

__device__ __forceinline__ f32x8 wmma_bf16(bf16x16 a, bf16x16 b, f32x8 c) {
    return __builtin_amdgcn_wmma_f32_16x16x32_bf16(false, a, false, b, (short)0, c, false, false);
}

// ---------------- fp32 -> bf16 conversion ----------------
__global__ __launch_bounds__(256) void k_f32_to_bf16(const float* __restrict__ in,
                                                     bf16* __restrict__ out, int n) {
    int i = blockIdx.x * 256 + threadIdx.x;
    if (i < n) out[i] = (bf16)in[i];
}

// ---------------- RMSNorm (one block per row) ----------------
__global__ __launch_bounds__(256) void k_rmsnorm(const float* __restrict__ X,
                                                 const float* __restrict__ G,
                                                 bf16* __restrict__ Out) {
    const int row = blockIdx.x;
    const float* x = X + (size_t)row * DM;
    float s = 0.f;
    #pragma unroll
    for (int i = threadIdx.x; i < DM; i += 256) { float v = x[i]; s += v * v; }
    #pragma unroll
    for (int m = 16; m >= 1; m >>= 1) s += __shfl_xor(s, m, 32);
    __shared__ float red[8];
    if ((threadIdx.x & 31) == 0) red[threadIdx.x >> 5] = s;
    __syncthreads();
    float tot = 0.f;
    #pragma unroll
    for (int j = 0; j < 8; ++j) tot += red[j];
    const float rinv = rsqrtf(tot / (float)DM + EPSV);
    bf16* o = Out + (size_t)row * DM;
    for (int i = threadIdx.x; i < DM; i += 256) o[i] = (bf16)(x[i] * rinv * G[i]);
}

// ---------------- WMMA NT-GEMM: C[M,N] = A[M,K] * B[N,K]^T ----------------
// Software-pipelined: tile k+1 is fetched into registers while tile k's WMMAs
// run; tile k+2 is prefetched into cache (global_prefetch_b8 on gfx1250).
// EPI 0: store f32          EPI 1: store f32 + R[m,n] (residual)
// EPI 2: store bf16( silu(R[m,n]) * acc )   (fused SwiGLU; R = w1 output)
template <int EPI>
__global__ __launch_bounds__(256)
void k_gemm_nt(const bf16* __restrict__ A, const bf16* __restrict__ B,
               const float* __restrict__ R, void* __restrict__ Out,
               int M, int N, int K) {
    constexpr int BM = 128, BN = 128, BK = 32, LDT = BK + 8;   // 40-elem stride
    __shared__ __attribute__((aligned(16))) bf16 As[BM * LDT];
    __shared__ __attribute__((aligned(16))) bf16 Bs[BN * LDT];

    const int m0 = blockIdx.y * BM, n0 = blockIdx.x * BN;
    const int tid  = threadIdx.x;
    const int lane = tid & 31, wid = tid >> 5;   // 8 waves
    const int wm = wid >> 2, wn = wid & 3;       // 2 x 4 wave grid: 64x32 per wave
    const int l15 = lane & 15, kh = lane >> 4;

    int rrow[2], rcol[2];
    #pragma unroll
    for (int i = 0; i < 2; ++i) {
        int idx = tid + i * 256;                 // 0..511 -> 128 rows x 32 cols
        rrow[i] = idx >> 2; rcol[i] = (idx & 3) * 8;
    }
    u32x4 ra[2], rb[2];
    auto fetch = [&](int k0) {
        #pragma unroll
        for (int i = 0; i < 2; ++i) {
            ra[i] = *(const u32x4*)(A + (size_t)(m0 + rrow[i]) * K + k0 + rcol[i]);
            rb[i] = *(const u32x4*)(B + (size_t)(n0 + rrow[i]) * K + k0 + rcol[i]);
        }
    };

    fetch(0);
    f32x8 acc[4][2] = {};

    for (int k0 = 0; k0 < K; k0 += BK) {
        #pragma unroll
        for (int i = 0; i < 2; ++i) {
            *(u32x4*)(As + rrow[i] * LDT + rcol[i]) = ra[i];
            *(u32x4*)(Bs + rrow[i] * LDT + rcol[i]) = rb[i];
        }
        __syncthreads();

        if (k0 + 2 * BK < K) {   // warm GL2/L0 two tiles ahead
            __builtin_prefetch(A + (size_t)(m0 + rrow[0]) * K + k0 + 2 * BK + rcol[0], 0, 3);
            __builtin_prefetch(B + (size_t)(n0 + rrow[0]) * K + k0 + 2 * BK + rcol[0], 0, 3);
        }
        if (k0 + BK < K) fetch(k0 + BK);   // loads in flight under the WMMAs

        bf16x16 afr[4], bfr[2];
        #pragma unroll
        for (int mi = 0; mi < 4; ++mi)
            afr[mi] = frag_ld(As + (wm * 64 + mi * 16 + l15) * LDT, kh);
        #pragma unroll
        for (int ni = 0; ni < 2; ++ni)
            bfr[ni] = frag_ld(Bs + (wn * 32 + ni * 16 + l15) * LDT, kh);
        #pragma unroll
        for (int mi = 0; mi < 4; ++mi)
            #pragma unroll
            for (int ni = 0; ni < 2; ++ni)
                acc[mi][ni] = wmma_bf16(afr[mi], bfr[ni], acc[mi][ni]);
        __syncthreads();
    }

    #pragma unroll
    for (int mi = 0; mi < 4; ++mi)
        #pragma unroll
        for (int ni = 0; ni < 2; ++ni)
            #pragma unroll
            for (int r = 0; r < 8; ++r) {
                int gm = m0 + wm * 64 + mi * 16 + kh * 8 + r;
                int gn = n0 + wn * 32 + ni * 16 + l15;
                size_t off = (size_t)gm * N + gn;
                float v = acc[mi][ni][r];
                if constexpr (EPI == 0) {
                    ((float*)Out)[off] = v;
                } else if constexpr (EPI == 1) {
                    ((float*)Out)[off] = v + R[off];
                } else {
                    float a = R[off];
                    float g = a / (1.f + __expf(-a));   // a * sigmoid(a)
                    ((bf16*)Out)[off] = (bf16)(g * v);
                }
            }
}

// ---------------- RoPE on Q,K + bf16 convert; V -> per-head transposed bf16 ----
// Q,K,V are fp32 (B,S,DM). Qb,Kb: bf16 (B,S,DM). Vt: bf16 [b][h][dk][S].
__global__ __launch_bounds__(256)
void k_rope(const float* __restrict__ Q, const float* __restrict__ K,
            const float* __restrict__ V, bf16* __restrict__ Qb,
            bf16* __restrict__ Kb, bf16* __restrict__ Vt) {
    const int pairsPerRow = DM / 2;                       // 512
    size_t idx = (size_t)blockIdx.x * 256 + threadIdx.x;  // over B*S*DM/2
    size_t row = idx / pairsPerRow;                       // b*S + s
    int pr = (int)(idx % pairsPerRow);
    int h = pr / (DK / 2);
    int p = pr % (DK / 2);
    int s = (int)(row % SEQ);
    int b = (int)(row / SEQ);
    int e0 = h * DK + 2 * p;

    float inv = __powf(THETA_F, -2.f * (float)p / (float)DK);
    float sn, cs;
    __sincosf((float)s * inv, &sn, &cs);

    size_t base = row * DM;
    float q1 = Q[base + e0], q2 = Q[base + e0 + 1];
    Qb[base + e0]     = (bf16)(q1 * cs - q2 * sn);
    Qb[base + e0 + 1] = (bf16)(q1 * sn + q2 * cs);
    float k1 = K[base + e0], k2 = K[base + e0 + 1];
    Kb[base + e0]     = (bf16)(k1 * cs - k2 * sn);
    Kb[base + e0 + 1] = (bf16)(k1 * sn + k2 * cs);

    float v1 = V[base + e0], v2 = V[base + e0 + 1];
    size_t vtb = ((size_t)b * HEADS + h) * DK;
    Vt[(vtb + 2 * p)     * SEQ + s] = (bf16)v1;
    Vt[(vtb + 2 * p + 1) * SEQ + s] = (bf16)v2;
}

// ---------------- Flash attention (causal, online softmax) ----------------
// Block: 128 threads (4 waves), handles one (b,h) and 64 query rows.
// Each wave owns a 16-row query strip; QK^T and P*V both via WMMA bf16.
// K/V tile kt+1 is fetched into registers while tile kt is being consumed.
__global__ __launch_bounds__(128)
void k_flash_attn(const bf16* __restrict__ Qb, const bf16* __restrict__ Kb,
                  const bf16* __restrict__ Vt, bf16* __restrict__ Ob) {
    constexpr int LT = 72;   // 64 + 8 pad (16B-aligned rows, conflict-free b128)
    __shared__ __attribute__((aligned(16))) bf16 Qs [64 * LT];
    __shared__ __attribute__((aligned(16))) bf16 Ks [64 * LT];
    __shared__ __attribute__((aligned(16))) bf16 Vts[64 * LT];
    __shared__ __attribute__((aligned(16))) bf16 Ps [64 * LT];

    const int qt = blockIdx.x;            // query tile (64 rows)
    const int bh = blockIdx.y;
    const int b = bh / HEADS, h = bh % HEADS;
    const int tid = threadIdx.x, lane = tid & 31, w = tid >> 5;
    const int l15 = lane & 15, kh = lane >> 4;

    const bf16* Qg = Qb + (size_t)b * SEQ * DM + h * DK;
    const bf16* Kg = Kb + (size_t)b * SEQ * DM + h * DK;
    const bf16* Vg = Vt + ((size_t)b * HEADS + h) * DK * SEQ;

    int trow[4], tcol[4];
    #pragma unroll
    for (int i = 0; i < 4; ++i) {          // 64x64 tile, 8 bf16 per thread-iter
        int idx = tid + i * 128;
        trow[i] = idx >> 3; tcol[i] = (idx & 7) * 8;
    }
    #pragma unroll
    for (int i = 0; i < 4; ++i)
        *(u32x4*)(Qs + trow[i] * LT + tcol[i]) =
            *(const u32x4*)(Qg + (size_t)(qt * 64 + trow[i]) * DM + tcol[i]);
    __syncthreads();

    const bf16* qrow = Qs + (w * 16 + l15) * LT;
    bf16x16 aq0 = frag_ld(qrow, kh), aq1 = frag_ld(qrow + 32, kh);

    float mrow[8], lrow[8];
    #pragma unroll
    for (int r = 0; r < 8; ++r) { mrow[r] = -1e30f; lrow[r] = 0.f; }
    f32x8 Oacc[4] = {};

    const int qg_base = qt * 64 + w * 16 + kh * 8;   // + r gives global q index

    u32x4 rk[4], rv[4];
    auto fetchKV = [&](int kt) {
        #pragma unroll
        for (int i = 0; i < 4; ++i) {
            rk[i] = *(const u32x4*)(Kg + (size_t)(kt * 64 + trow[i]) * DM + tcol[i]);
            rv[i] = *(const u32x4*)(Vg + (size_t)trow[i] * SEQ + kt * 64 + tcol[i]);
        }
    };
    fetchKV(0);

    for (int kt = 0; kt <= qt; ++kt) {
        __syncthreads();   // previous-iteration readers of Ks/Vts are done
        #pragma unroll
        for (int i = 0; i < 4; ++i) {
            *(u32x4*)(Ks  + trow[i] * LT + tcol[i]) = rk[i];
            *(u32x4*)(Vts + trow[i] * LT + tcol[i]) = rv[i];
        }
        __syncthreads();

        if (kt + 2 <= qt) {   // warm GL2 two tiles ahead
            __builtin_prefetch(Kg + (size_t)((kt + 2) * 64 + trow[0]) * DM + tcol[0], 0, 3);
            __builtin_prefetch(Vg + (size_t)trow[0] * SEQ + (kt + 2) * 64 + tcol[0], 0, 3);
        }
        if (kt < qt) fetchKV(kt + 1);   // loads in flight under score/softmax

        // scores: 16(q) x 64(k) per wave, K-dim = dk = 64 -> 2 chained WMMAs
        f32x8 st[4];
        #pragma unroll
        for (int nt = 0; nt < 4; ++nt) {
            const bf16* krow = Ks + (nt * 16 + l15) * LT;
            f32x8 a = {};
            a = wmma_bf16(aq0, frag_ld(krow, kh), a);
            a = wmma_bf16(aq1, frag_ld(krow + 32, kh), a);
            st[nt] = a;
        }
        // scale + causal mask
        #pragma unroll
        for (int nt = 0; nt < 4; ++nt) {
            int kg = kt * 64 + nt * 16 + l15;
            #pragma unroll
            for (int r = 0; r < 8; ++r) {
                float v = st[nt][r] * 0.125f;                // 1/sqrt(64)
                st[nt][r] = (kg > qg_base + r) ? -1e30f : v;
            }
        }
        // online softmax: row max over 16 lanes (columns) then over nt
        float sc[8];
        #pragma unroll
        for (int r = 0; r < 8; ++r) {
            float mx = fmaxf(fmaxf(st[0][r], st[1][r]), fmaxf(st[2][r], st[3][r]));
            #pragma unroll
            for (int m = 1; m < 16; m <<= 1) mx = fmaxf(mx, __shfl_xor(mx, m, 32));
            float mnew = fmaxf(mrow[r], mx);
            sc[r] = __expf(mrow[r] - mnew);
            #pragma unroll
            for (int nt = 0; nt < 4; ++nt) st[nt][r] = __expf(st[nt][r] - mnew);
            float sm = st[0][r] + st[1][r] + st[2][r] + st[3][r];
            #pragma unroll
            for (int m = 1; m < 16; m <<= 1) sm += __shfl_xor(sm, m, 32);
            lrow[r] = lrow[r] * sc[r] + sm;
            mrow[r] = mnew;
        }
        #pragma unroll
        for (int nt = 0; nt < 4; ++nt)
            #pragma unroll
            for (int r = 0; r < 8; ++r) Oacc[nt][r] *= sc[r];

        // P: C-layout -> LDS -> A-layout fragments
        #pragma unroll
        for (int nt = 0; nt < 4; ++nt)
            #pragma unroll
            for (int r = 0; r < 8; ++r)
                Ps[(w * 16 + kh * 8 + r) * LT + nt * 16 + l15] = (bf16)st[nt][r];
        __syncthreads();

        const bf16* prow = Ps + (w * 16 + l15) * LT;
        bf16x16 ap0 = frag_ld(prow, kh), ap1 = frag_ld(prow + 32, kh);
        #pragma unroll
        for (int nt = 0; nt < 4; ++nt) {
            const bf16* vrow = Vts + (nt * 16 + l15) * LT;
            Oacc[nt] = wmma_bf16(ap0, frag_ld(vrow, kh), Oacc[nt]);
            Oacc[nt] = wmma_bf16(ap1, frag_ld(vrow + 32, kh), Oacc[nt]);
        }
    }

    // normalize and write bf16 output (feeds the WO GEMM directly)
    #pragma unroll
    for (int nt = 0; nt < 4; ++nt)
        #pragma unroll
        for (int r = 0; r < 8; ++r) {
            float o = Oacc[nt][r] / lrow[r];
            size_t srow = (size_t)b * SEQ + qt * 64 + w * 16 + kh * 8 + r;
            Ob[srow * DM + h * DK + nt * 16 + l15] = (bf16)o;
        }
}

// ---------------- host launcher ----------------
extern "C" void kernel_launch(void* const* d_in, const int* in_sizes, int n_in,
                              void* d_out, int out_size, void* d_ws, size_t ws_size,
                              hipStream_t stream) {
    (void)in_sizes; (void)n_in; (void)out_size; (void)ws_size;
    const float* x  = (const float*)d_in[0];
    const float* wq = (const float*)d_in[1];
    const float* wk = (const float*)d_in[2];
    const float* wv = (const float*)d_in[3];
    const float* wo = (const float*)d_in[4];
    const float* w1 = (const float*)d_in[5];
    const float* w2 = (const float*)d_in[6];
    const float* w3 = (const float*)d_in[7];
    const float* g1 = (const float*)d_in[8];
    const float* g2 = (const float*)d_in[9];
    float* out = (float*)d_out;

    const size_t MB = 1ull << 20;
    char* p = (char*)d_ws;
    bf16*  n_bf  = (bf16*)p;            p += 8 * MB;   // 4096x1024 bf16
    bf16*  wq_bf = (bf16*)p;            p += 2 * MB;
    bf16*  wk_bf = (bf16*)p;            p += 2 * MB;
    bf16*  wv_bf = (bf16*)p;            p += 2 * MB;
    bf16*  wo_bf = (bf16*)p;            p += 2 * MB;
    bf16*  w1_bf = (bf16*)p;            p += 8 * MB;
    bf16*  w3_bf = (bf16*)p;            p += 8 * MB;
    bf16*  w2_bf = (bf16*)p;            p += 8 * MB;
    float* x1    = (float*)p;           p += 16 * MB;  // x + attn residual
    bf16*  n2_bf = (bf16*)p;            p += 8 * MB;
    // 64 MB region reused: {q_f,k_f,v_f fp32 + q_bf,k_bf} then a_f (w1 output)
    char*  reuse = p;                   p += 64 * MB;
    float* q_f  = (float*)reuse;
    float* k_f  = (float*)(reuse + 16 * MB);
    float* v_f  = (float*)(reuse + 32 * MB);
    bf16*  q_bf = (bf16*) (reuse + 48 * MB);
    bf16*  k_bf = (bf16*) (reuse + 56 * MB);
    float* a_f  = (float*)reuse;                       // 64 MB, after attention
    bf16*  vt_bf = (bf16*)p;            p += 8 * MB;
    bf16*  o_bf  = (bf16*)p;            p += 8 * MB;
    bf16*  h_bf  = (bf16*)p;            p += 32 * MB;  // swiglu output 4096x4096

    // 1) weights -> bf16
    auto conv = [&](const float* src, bf16* dst, int n) {
        k_f32_to_bf16<<<(n + 255) / 256, 256, 0, stream>>>(src, dst, n);
    };
    conv(wq, wq_bf, DM * DM);  conv(wk, wk_bf, DM * DM);
    conv(wv, wv_bf, DM * DM);  conv(wo, wo_bf, DM * DM);
    conv(w1, w1_bf, DFF * DM); conv(w3, w3_bf, DFF * DM); conv(w2, w2_bf, DM * DFF);

    // 2) n = rmsnorm(x, g1)
    k_rmsnorm<<<MROWS, 256, 0, stream>>>(x, g1, n_bf);

    // 3) Q,K,V projections (fp32 out, then RoPE)
    dim3 gQKV(DM / 128, MROWS / 128);
    k_gemm_nt<0><<<gQKV, 256, 0, stream>>>(n_bf, wq_bf, nullptr, q_f, MROWS, DM, DM);
    k_gemm_nt<0><<<gQKV, 256, 0, stream>>>(n_bf, wk_bf, nullptr, k_f, MROWS, DM, DM);
    k_gemm_nt<0><<<gQKV, 256, 0, stream>>>(n_bf, wv_bf, nullptr, v_f, MROWS, DM, DM);

    // 4) RoPE(q,k) -> bf16; V -> per-head transposed bf16
    k_rope<<<(BATCH * SEQ * DM / 2) / 256, 256, 0, stream>>>(q_f, k_f, v_f,
                                                             q_bf, k_bf, vt_bf);

    // 5) causal flash attention -> o (bf16)
    dim3 gAtt(SEQ / 64, BATCH * HEADS);
    k_flash_attn<<<gAtt, 128, 0, stream>>>(q_bf, k_bf, vt_bf, o_bf);

    // 6) x1 = x + o @ Wo^T
    k_gemm_nt<1><<<gQKV, 256, 0, stream>>>(o_bf, wo_bf, x, x1, MROWS, DM, DM);

    // 7) n2 = rmsnorm(x1, g2)
    k_rmsnorm<<<MROWS, 256, 0, stream>>>(x1, g2, n2_bf);

    // 8) a = n2 @ W1^T ; h = silu(a) * (n2 @ W3^T)  [fused epilogue]
    dim3 gFF(DFF / 128, MROWS / 128);
    k_gemm_nt<0><<<gFF, 256, 0, stream>>>(n2_bf, w1_bf, nullptr, a_f, MROWS, DFF, DM);
    k_gemm_nt<2><<<gFF, 256, 0, stream>>>(n2_bf, w3_bf, a_f, h_bf, MROWS, DFF, DM);

    // 9) out = x1 + h @ W2^T
    k_gemm_nt<1><<<gQKV, 256, 0, stream>>>(h_bf, w2_bf, x1, out, MROWS, DM, DFF);
}
